// ConditioningEncoder_29394756173827
// MI455X (gfx1250) — matrix operand order repaired and checked by reading, not compiled, not run on hardware
//
#include <hip/hip_runtime.h>
#include <hip/hip_bf16.h>

// ---------------------------------------------------------------------------
// ConditioningEncoder for gfx1250 (MI455X).
// Heavy convs: WMMA GEMMs  D[16x16 f32] += A[16x32 f16] x B[32x16 f16]
// (v_wmma_f32_16x16x32_f16, wave32). Input tiles staged to LDS with
// GLOBAL_LOAD_ASYNC_TO_LDS (ASYNCcnt) when the builtin is available.
// ---------------------------------------------------------------------------

typedef __attribute__((ext_vector_type(16))) _Float16 v16h;
typedef __attribute__((ext_vector_type(8)))  _Float16 v8h;
typedef __attribute__((ext_vector_type(8)))  float    v8f;

#define NB 8        // batch
#define NP 64       // phonemes
#define NT 1024     // frames
#define NH 256      // hidden
#define NE 256      // embed
#define NHH 128     // hidden/2
#define NFREQ 80

#define MTILE 128   // Cout tile per block (each wave: 2 x 16 rows)
#define NTILE 32    // time tile per block
#define KCH   32    // K chunk (matches WMMA K)

#if defined(__gfx1250__) && __has_builtin(__builtin_amdgcn_global_load_async_to_lds_b32)
#define HAVE_ASYNC_LDS 1
typedef __attribute__((address_space(1))) int* gas_i32p;
typedef __attribute__((address_space(3))) int* las_i32p;
#else
#define HAVE_ASYNC_LDS 0
#endif

__device__ __forceinline__ void async_wait_all()
{
#if defined(__gfx1250__)
#if __has_builtin(__builtin_amdgcn_s_wait_asynccnt)
    __builtin_amdgcn_s_wait_asynccnt(0);
#else
    asm volatile("s_wait_asynccnt 0x0" ::: "memory");
#endif
#endif
}

// ---------------------------------------------------------------------------
// Generic conv1d (kernel size KSZ, pad KSZ/2) as WMMA GEMM.
// in  : [B, CIN, L]  f32
// w   : [COUT, CIN, KSZ] f32  (== [COUT, K], K = CIN*KSZ, ci-major)
// out : [B, COUT, L] f32 (pre-BN, bias added)
// grid = (L/NTILE, COUT/MTILE, B), block = 256 (8 wave32).
// Wave (wm,wn): rows [wm*32, wm*32+32) x cols [wn*16, wn*16+16); two WMMAs
// per K-chunk sharing one B fragment.
// ---------------------------------------------------------------------------
template<int CIN, int COUT, int KSZ>
__global__ __launch_bounds__(256)
void conv1d_wmma(const float* __restrict__ in, const float* __restrict__ w,
                 const float* __restrict__ bias, float* __restrict__ out, int L)
{
    constexpr int PAD = KSZ / 2;
    constexpr int NTH = NTILE + KSZ - 1;   // staged columns (with halo)
    constexpr int K   = CIN * KSZ;

    __shared__ float sInF[CIN * NTH];                                  // raw f32 tile
    __shared__ __attribute__((aligned(32))) _Float16 sW[MTILE * KCH];  // weight chunk
    __shared__ __attribute__((aligned(32))) _Float16 sB[NTILE * KCH];  // im2col chunk (K-major per col)

    const int b      = blockIdx.z;
    const int coBase = blockIdx.y * MTILE;
    const int tBase  = blockIdx.x * NTILE;
    const int tid    = threadIdx.x;
    const int lane   = tid & 31;
    const int wave   = tid >> 5;
    const int wm     = wave >> 1;   // 0..3 -> 32-row slice of 128
    const int wn     = wave & 1;    // 0..1 -> 16-col slice of 32

    // ---- Stage input tile once (async DMA to LDS when available) ----
    for (int i = tid; i < CIN * NTH; i += 256) {
        int ci = i / NTH, j = i - ci * NTH;
        int t  = tBase - PAD + j;
        if (t >= 0 && t < L) {
#if HAVE_ASYNC_LDS
            __builtin_amdgcn_global_load_async_to_lds_b32(
                (gas_i32p)(in + ((size_t)b * CIN + ci) * L + t),
                (las_i32p)(&sInF[i]), 0, 0);
#else
            sInF[i] = in[((size_t)b * CIN + ci) * L + t];
#endif
        } else {
            sInF[i] = 0.0f;
        }
    }
#if HAVE_ASYNC_LDS
    async_wait_all();
#endif

    union AF { v16h v; v8h  h8[2]; };
    union CF { v8f  v; float f[8]; };
    CF acc0, acc1;
#pragma unroll
    for (int r = 0; r < 8; ++r) { acc0.f[r] = 0.0f; acc1.f[r] = 0.0f; }

    // WMMA operand k order used here (wave32, contiguous; A and B match):
    //  A 16x32 f16: M = lane%16; K = kbA+0..15,              kbA = lane<16?0:16
    //  B 32x16 f16: N = lane%16; K = kbB+0..15,              kbB = lane<16?0:16
    const int kbA  = (lane < 16) ? 0 : 16;
    const int kbB  = (lane < 16) ? 0 : 16;
    const int mrow = wm * 32 + (lane & 15);
    const int ncol = wn * 16 + (lane & 15);

    for (int kc = 0; kc < K; kc += KCH) {
        __syncthreads();
        __builtin_prefetch(&w[(size_t)coBase * K + kc + KCH], 0, 1); // global_prefetch_b8
        // Stage 128x32 weight chunk (f32 -> f16), row-major contiguous K.
        for (int i = tid; i < MTILE * KCH; i += 256) {
            int m = i >> 5, kkl = i & 31;
            sW[i] = (_Float16)w[(size_t)(coBase + m) * K + kc + kkl];
        }
        // Build K-major im2col chunk so each lane's B fragment is contiguous.
        for (int i = tid; i < NTILE * KCH; i += 256) {
            int n = i >> 5, kkl = i & 31;
            int kkg = kc + kkl;
            int ci  = kkg / KSZ, kk = kkg - ci * KSZ;
            sB[i] = (_Float16)sInF[ci * NTH + n + kk];
        }
        __syncthreads();

        AF a0, a1;
        a0.h8[0] = *(const v8h*)(sW + mrow * KCH + kbA);
        a0.h8[1] = *(const v8h*)(sW + mrow * KCH + kbA + 8);
        a1.h8[0] = *(const v8h*)(sW + (mrow + 16) * KCH + kbA);
        a1.h8[1] = *(const v8h*)(sW + (mrow + 16) * KCH + kbA + 8);
        v16h bf  = *(const v16h*)(sB + ncol * KCH + kbB);   // 2 x ds_load_b128

        acc0.v = __builtin_amdgcn_wmma_f32_16x16x32_f16(
            false, a0.v, false, bf, (short)0, acc0.v, false, false);
        acc1.v = __builtin_amdgcn_wmma_f32_16x16x32_f16(
            false, a1.v, false, bf, (short)0, acc1.v, false, false);
    }

    // C/D layout: VGPR r -> M = r (lanes 0-15) or r+8 (lanes 16-31); N = lane%16.
    const int co0 = coBase + wm * 32 + ((lane < 16) ? 0 : 8);
    const int t   = tBase + ncol;
#pragma unroll
    for (int r = 0; r < 8; ++r) {
        out[((size_t)b * COUT + (co0 + r)) * L + t]      = acc0.f[r] + bias[co0 + r];
        out[((size_t)b * COUT + (co0 + 16 + r)) * L + t] = acc1.f[r] + bias[co0 + 16 + r];
    }
}

// ---- Cin==1, k=3 conv (f0 branch layer 1): plain VALU --------------------
__global__ void conv1d_c1(const float* __restrict__ in, const float* __restrict__ w,
                          const float* __restrict__ bias, float* __restrict__ out,
                          int Cout, int L, int total)
{
    int i = blockIdx.x * 256 + threadIdx.x;
    if (i >= total) return;
    int t = i % L, c = (i / L) % Cout, b = i / (L * Cout);
    const float* xb = in + (size_t)b * L;
    float acc = bias[c];
#pragma unroll
    for (int k = 0; k < 3; ++k) {
        int tt = t + k - 1;
        if (tt >= 0 && tt < L) acc += w[c * 3 + k] * xb[tt];
    }
    out[i] = acc;
}

// ---- BatchNorm (training-mode batch stats over N,L): stats then apply ----
__global__ __launch_bounds__(256)
void bn_stats(const float* __restrict__ x, int C, int L,
              float* __restrict__ mean, float* __restrict__ istd)
{
    int c = blockIdx.x, tid = threadIdx.x;
    const int N = NB * L;
    float s = 0.f, s2 = 0.f;
    for (int i = tid; i < N; i += 256) {
        int b = i / L, l = i - b * L;
        float v = x[((size_t)b * C + c) * L + l];
        s += v; s2 += v * v;
    }
    __shared__ float sh1[256], sh2[256];
    sh1[tid] = s; sh2[tid] = s2;
    __syncthreads();
    for (int o = 128; o > 0; o >>= 1) {
        if (tid < o) { sh1[tid] += sh1[tid + o]; sh2[tid] += sh2[tid + o]; }
        __syncthreads();
    }
    if (tid == 0) {
        float m = sh1[0] / (float)N;
        float v = sh2[0] / (float)N - m * m;
        mean[c] = m;
        istd[c] = rsqrtf(v + 1e-5f);
    }
}

// mode 0 = relu, 1 = leaky(0.1). Writes into channel slot cOff of an
// outC-channel destination tensor (free concatenation).
__global__ void bn_act(const float* __restrict__ x, const float* __restrict__ mean,
                       const float* __restrict__ istd, const float* __restrict__ g,
                       const float* __restrict__ be, float* __restrict__ outB,
                       int C, int L, int outC, int cOff, int mode, int total)
{
    int i = blockIdx.x * 256 + threadIdx.x;
    if (i >= total) return;
    int l = i % L, c = (i / L) % C, b = i / (L * C);
    float v = (x[i] - mean[c]) * istd[c] * g[c] + be[c];
    v = (mode == 0) ? fmaxf(v, 0.f) : ((v > 0.f) ? v : 0.1f * v);
    outB[((size_t)b * outC + cOff + c) * L + l] = v;
}

// ---- Gathers / pointwise --------------------------------------------------
__global__ void ph_gather(const int* __restrict__ ids, const float* __restrict__ emb,
                          float* __restrict__ out, int total)
{   // out [B, E, P]
    int i = blockIdx.x * 256 + threadIdx.x;
    if (i >= total) return;
    int p = i % NP, e = (i / NP) % NE, b = i / (NP * NE);
    out[i] = emb[(size_t)ids[b * NP + p] * NE + e];
}

__global__ void pt_gather(const int* __restrict__ midi, const float* __restrict__ emb,
                          float* __restrict__ outB, int outC, int total)
{   // writes [B, H, T] into channel slot 0 of outC-channel tensor
    int i = blockIdx.x * 256 + threadIdx.x;
    if (i >= total) return;
    int t = i % NT, h = (i / NT) % NH, b = i / (NT * NH);
    int m = midi[b * NT + t];
    m = min(max(m, 0), 127);
    outB[((size_t)b * outC + h) * NT + t] = emb[(size_t)m * NH + h];
}

__global__ void f0_norm(const float* __restrict__ f0, float* __restrict__ out, int total)
{
    int i = blockIdx.x * 256 + threadIdx.x;
    if (i >= total) return;
    float v = f0[i];
    out[i] = (v > 0.f) ? v * (1.0f / 500.0f) : 0.f;
}

__global__ void cumsum_dur(const int* __restrict__ dur, int* __restrict__ cum)
{
    int b = blockIdx.x;
    if (threadIdx.x == 0) {
        int s = 0;
        for (int p = 0; p < NP; ++p) { s += max(dur[b * NP + p], 0); cum[b * NP + p] = s; }
    }
}

__global__ void frame_idx(const int* __restrict__ cum, int* __restrict__ idxv, int total)
{   // searchsorted(cum, t, side='right'); -1 marks invalid (t >= total duration)
    int i = blockIdx.x * 256 + threadIdx.x;
    if (i >= total) return;
    int t = i % NT, b = i / NT;
    const int* cb = cum + b * NP;
    int idx = 0;
    while (idx < NP && cb[idx] <= t) ++idx;
    if (t >= cb[NP - 1]) { idxv[i] = -1; return; }
    idxv[i] = min(idx, NP - 1);
}

__global__ void expand_lr(const float* __restrict__ xb, const int* __restrict__ idxv,
                          float* __restrict__ outB, int outC, int total)
{   // xb [B, H, P] -> outB channel slot 0 of [B, outC, T]
    int i = blockIdx.x * 256 + threadIdx.x;
    if (i >= total) return;
    int t = i % NT, h = (i / NT) % NH, b = i / (NT * NH);
    int id = idxv[b * NT + t];
    float v = (id >= 0) ? xb[((size_t)b * NH + h) * NP + id] : 0.f;
    outB[((size_t)b * outC + h) * NT + t] = v;
}

// ---- Final bilinear upsample (half-pixel centers, 2 -> 80 on freq axis) --
__global__ __launch_bounds__(256)
void resize_bilinear(const float* __restrict__ fp, float* __restrict__ out, long long total)
{   // fp [B, 2H, T]; ff[b,h,j,t] = fp[b, j*H+h, t]; out [B, H, FREQ, T]
    long long i = (long long)blockIdx.x * 256 + threadIdx.x;
    if (i >= total) return;
    int t = (int)(i % NT);
    int r = (int)((i / NT) % NFREQ);
    int h = (int)((i / ((long long)NT * NFREQ)) % NH);
    int b = (int)(i / ((long long)NT * NFREQ * NH));
    float src = (r + 0.5f) * (2.0f / (float)NFREQ) - 0.5f;
    float fl  = floorf(src);
    int   i0  = (int)fl;
    float f   = src - fl;
    int i0c = min(max(i0, 0), 1);
    int i1c = min(max(i0 + 1, 0), 1);
    float v0 = fp[((size_t)b * (2 * NH) + i0c * NH + h) * NT + t];
    float v1 = fp[((size_t)b * (2 * NH) + i1c * NH + h) * NT + t];
    out[i] = (1.f - f) * v0 + f * v1;
}

// ---------------------------------------------------------------------------
static inline int gdiv(int a, int b) { return (a + b - 1) / b; }

extern "C" void kernel_launch(void* const* d_in, const int* in_sizes, int n_in,
                              void* d_out, int out_size, void* d_ws, size_t ws_size,
                              hipStream_t stream)
{
    // Inputs (setup_inputs dict order; params dict flattened in insertion order)
    const int*   phoneme_ids = (const int*)d_in[0];
    const int*   durations   = (const int*)d_in[1];
    const int*   midi        = (const int*)d_in[2];
    const float* f0          = (const float*)d_in[3];
    // d_in[4] = mel_shape (FREQ, T) — compile-time constants here
    const float* ph_emb = (const float*)d_in[5];
    const float* ph_w1  = (const float*)d_in[6];
    const float* ph_b1  = (const float*)d_in[7];
    const float* ph_g1  = (const float*)d_in[8];
    const float* ph_be1 = (const float*)d_in[9];
    const float* ph_w2  = (const float*)d_in[10];
    const float* ph_b2  = (const float*)d_in[11];
    const float* ph_g2  = (const float*)d_in[12];
    const float* ph_be2 = (const float*)d_in[13];
    const float* pt_emb = (const float*)d_in[14];
    const float* f0_w1  = (const float*)d_in[15];
    const float* f0_b1  = (const float*)d_in[16];
    const float* f0_g1  = (const float*)d_in[17];
    const float* f0_be1 = (const float*)d_in[18];
    const float* f0_w2  = (const float*)d_in[19];
    const float* f0_b2  = (const float*)d_in[20];
    const float* f0_g2  = (const float*)d_in[21];
    const float* f0_be2 = (const float*)d_in[22];
    const float* pc_w   = (const float*)d_in[23];
    const float* pc_b   = (const float*)d_in[24];
    const float* pc_g   = (const float*)d_in[25];
    const float* pc_be  = (const float*)d_in[26];
    const float* cb_w1  = (const float*)d_in[27];
    const float* cb_b1  = (const float*)d_in[28];
    const float* cb_g1  = (const float*)d_in[29];
    const float* cb_be1 = (const float*)d_in[30];
    const float* cb_w2  = (const float*)d_in[31];
    const float* cb_b2  = (const float*)d_in[32];
    const float* cb_g2  = (const float*)d_in[33];
    const float* cb_be2 = (const float*)d_in[34];
    const float* fp_w   = (const float*)d_in[35];
    const float* fp_b   = (const float*)d_in[36];

    float* outp = (float*)d_out;

    // Workspace layout (~17.5M floats ~= 70 MB)
    float* wsf = (float*)d_ws;
    size_t off = 0;
    auto alloc = [&](size_t n) { float* p = wsf + off; off += n; return p; };
    float* mean   = alloc(512);
    float* istd   = alloc(512);
    int*   cum    = (int*)alloc(NB * NP);
    int*   idxv   = (int*)alloc(NB * NT);
    float* x0     = alloc((size_t)NB * NE * NP);
    float* ph1raw = alloc((size_t)NB * NH * NP);
    float* ph1act = alloc((size_t)NB * NH * NP);
    float* ph2raw = alloc((size_t)NB * NH * NP);
    float* ph2act = alloc((size_t)NB * NH * NP);
    float* f0nb   = alloc((size_t)NB * NT);
    float* catA   = alloc((size_t)NB * 2 * NH * NT);   // [pf | g] -> pc conv input
    float* catC   = alloc((size_t)NB * 2 * NH * NT);   // [x  | pitch] -> cb conv input
    float* r3     = alloc((size_t)NB * 2 * NH * NT);   // multi-use scratch
    float* fpout  = alloc((size_t)NB * 2 * NH * NT);
    float* g1raw    = r3;
    float* g1act    = r3 + (size_t)NB * NHH * NT;
    float* g2raw    = r3 + 2 * (size_t)NB * NHH * NT;
    float* pitchraw = r3;                     // reuse after g-branch is consumed
    float* tmp1     = r3;
    float* tmp2     = r3 + (size_t)NB * NH * NT;

    // ---- Phoneme encoder ----
    {
        int tot = NB * NE * NP;
        ph_gather<<<gdiv(tot, 256), 256, 0, stream>>>(phoneme_ids, ph_emb, x0, tot);
    }
    conv1d_wmma<NE, NH, 3><<<dim3(NP / NTILE, NH / MTILE, NB), 256, 0, stream>>>(x0, ph_w1, ph_b1, ph1raw, NP);
    bn_stats<<<NH, 256, 0, stream>>>(ph1raw, NH, NP, mean, istd);
    {
        int tot = NB * NH * NP;
        bn_act<<<gdiv(tot, 256), 256, 0, stream>>>(ph1raw, mean, istd, ph_g1, ph_be1, ph1act, NH, NP, NH, 0, 0, tot);
    }
    conv1d_wmma<NH, NH, 3><<<dim3(NP / NTILE, NH / MTILE, NB), 256, 0, stream>>>(ph1act, ph_w2, ph_b2, ph2raw, NP);
    bn_stats<<<NH, 256, 0, stream>>>(ph2raw, NH, NP, mean, istd);
    {
        int tot = NB * NH * NP;
        bn_act<<<gdiv(tot, 256), 256, 0, stream>>>(ph2raw, mean, istd, ph_g2, ph_be2, ph2act, NH, NP, NH, 0, 0, tot);
    }

    // ---- Length regulator: ph2act [B,H,P] -> catC channels 0..H-1 ----
    cumsum_dur<<<NB, 32, 0, stream>>>(durations, cum);
    {
        int tot = NB * NT;
        frame_idx<<<gdiv(tot, 256), 256, 0, stream>>>(cum, idxv, tot);
    }
    {
        int tot = NB * NH * NT;
        expand_lr<<<gdiv(tot, 256), 256, 0, stream>>>(ph2act, idxv, catC, 2 * NH, tot);
    }

    // ---- Pitch encoder ----
    {
        int tot = NB * NH * NT;
        pt_gather<<<gdiv(tot, 256), 256, 0, stream>>>(midi, pt_emb, catA, 2 * NH, tot);  // pf -> catA[0..H)
    }
    {
        int tot = NB * NT;
        f0_norm<<<gdiv(tot, 256), 256, 0, stream>>>(f0, f0nb, tot);
    }
    {
        int tot = NB * NHH * NT;
        conv1d_c1<<<gdiv(tot, 256), 256, 0, stream>>>(f0nb, f0_w1, f0_b1, g1raw, NHH, NT, tot);
        bn_stats<<<NHH, 256, 0, stream>>>(g1raw, NHH, NT, mean, istd);
        bn_act<<<gdiv(tot, 256), 256, 0, stream>>>(g1raw, mean, istd, f0_g1, f0_be1, g1act, NHH, NT, NHH, 0, 1, tot);
    }
    conv1d_wmma<NHH, NH, 3><<<dim3(NT / NTILE, NH / MTILE, NB), 256, 0, stream>>>(g1act, f0_w2, f0_b2, g2raw, NT);
    bn_stats<<<NH, 256, 0, stream>>>(g2raw, NH, NT, mean, istd);
    {
        int tot = NB * NH * NT;
        bn_act<<<gdiv(tot, 256), 256, 0, stream>>>(g2raw, mean, istd, f0_g2, f0_be2, catA, NH, NT, 2 * NH, NH, 1, tot);  // g -> catA[H..2H)
    }
    conv1d_wmma<2 * NH, NH, 3><<<dim3(NT / NTILE, NH / MTILE, NB), 256, 0, stream>>>(catA, pc_w, pc_b, pitchraw, NT);
    bn_stats<<<NH, 256, 0, stream>>>(pitchraw, NH, NT, mean, istd);
    {
        int tot = NB * NH * NT;
        bn_act<<<gdiv(tot, 256), 256, 0, stream>>>(pitchraw, mean, istd, pc_g, pc_be, catC, NH, NT, 2 * NH, NH, 1, tot); // pitch -> catC[H..2H)
    }

    // ---- Combiner ----
    conv1d_wmma<2 * NH, NH, 3><<<dim3(NT / NTILE, NH / MTILE, NB), 256, 0, stream>>>(catC, cb_w1, cb_b1, tmp1, NT);
    bn_stats<<<NH, 256, 0, stream>>>(tmp1, NH, NT, mean, istd);
    {
        int tot = NB * NH * NT;
        bn_act<<<gdiv(tot, 256), 256, 0, stream>>>(tmp1, mean, istd, cb_g1, cb_be1, tmp2, NH, NT, NH, 0, 1, tot);
    }
    conv1d_wmma<NH, NH, 3><<<dim3(NT / NTILE, NH / MTILE, NB), 256, 0, stream>>>(tmp2, cb_w2, cb_b2, tmp1, NT);
    bn_stats<<<NH, 256, 0, stream>>>(tmp1, NH, NT, mean, istd);
    {
        int tot = NB * NH * NT;
        bn_act<<<gdiv(tot, 256), 256, 0, stream>>>(tmp1, mean, istd, cb_g2, cb_be2, tmp2, NH, NT, NH, 0, 1, tot);
    }

    // ---- Frequency projection (1x1 conv, also WMMA) ----
    conv1d_wmma<NH, 2 * NH, 1><<<dim3(NT / NTILE, (2 * NH) / MTILE, NB), 256, 0, stream>>>(tmp2, fp_w, fp_b, fpout, NT);

    // ---- Bilinear upsample 2 -> 80 (half-pixel centers) into d_out ----
    {
        long long tot = (long long)NB * NH * NFREQ * NT;
        resize_bilinear<<<(unsigned)((tot + 255) / 256), 256, 0, stream>>>(fpout, outp, tot);
    }
}
